// OAMatcherNet_58248346468588
// MI455X (gfx1250) — matrix operand order repaired and checked by reading, not compiled, not run on hardware
//
#include <hip/hip_runtime.h>
#include <hip/hip_bf16.h>
#include <math.h>

// ---------------- problem constants ----------------
constexpr int Bn  = 4;
constexpr int Nn  = 1024;
constexpr int KNN = 20;
constexpr int Dn  = 512;
constexpr int Hn  = 4;
constexpr int FFn = 1024;
constexpr int DKn = Dn / Hn;          // 128
constexpr int BNr = Bn * Nn;          // 4096 rows
constexpr long MBIG = (long)Bn * Nn * KNN;   // 81920 rows for edge-conv GEMMs

typedef __attribute__((ext_vector_type(16))) __bf16 bf16x16;
typedef __attribute__((ext_vector_type(8)))  float  f32x8;

// ---------------- WMMA fragment loaders (f32 -> bf16 inline) ----------------
// A frag (16x32, lane half hi): lanes0-15 hold K kA..kA+7 (v0-3) & kA+16..kA+23 (v4-7)
__device__ __forceinline__ bf16x16 load_fragA_fast(const float* __restrict__ ap, int kA)
{
  bf16x16 a;
  float t[16];
  const float4* A4 = reinterpret_cast<const float4*>(ap + kA);
  *reinterpret_cast<float4*>(t + 0)  = A4[0];
  *reinterpret_cast<float4*>(t + 4)  = A4[1];
  *reinterpret_cast<float4*>(t + 8)  = A4[4];   // +16 floats
  *reinterpret_cast<float4*>(t + 12) = A4[5];
#pragma unroll
  for (int i = 0; i < 16; ++i) a[i] = (__bf16)t[i];
  return a;
}
__device__ __forceinline__ bf16x16 load_fragA_guard(const float* __restrict__ ap, int kA, int K)
{
  bf16x16 a;
#pragma unroll
  for (int i = 0; i < 8; ++i) {
    int k1 = kA + i;      a[i]     = (__bf16)((k1 < K) ? ap[k1] : 0.f);
    int k2 = kA + 16 + i; a[8 + i] = (__bf16)((k2 < K) ? ap[k2] : 0.f);
  }
  return a;
}
// B frag (32x16): lane = N col, K kB..kB+7 & kB+16..kB+23
__device__ __forceinline__ bf16x16 load_fragB_fast(const float* __restrict__ bp, int kB)
{
  bf16x16 b;
  float t[16];
  const float4* B4 = reinterpret_cast<const float4*>(bp + kB);
  *reinterpret_cast<float4*>(t + 0)  = B4[0];
  *reinterpret_cast<float4*>(t + 4)  = B4[1];
  *reinterpret_cast<float4*>(t + 8)  = B4[4];
  *reinterpret_cast<float4*>(t + 12) = B4[5];
#pragma unroll
  for (int i = 0; i < 16; ++i) b[i] = (__bf16)t[i];
  return b;
}
__device__ __forceinline__ bf16x16 load_fragB_guard(const float* __restrict__ bp, int kB, int K)
{
  bf16x16 b;
#pragma unroll
  for (int i = 0; i < 8; ++i) {
    int k1 = kB + i;      b[i]     = (__bf16)((k1 < K) ? bp[k1] : 0.f);
    int k2 = kB + 16 + i; b[8 + i] = (__bf16)((k2 < K) ? bp[k2] : 0.f);
  }
  return b;
}

// =====================================================================
// Generic NT GEMM on WMMA:  C[M,N] = alpha * A[M,K] @ B[N,K]^T (+bias)(+relu)
// One wave computes a 32x32 C macro-tile (2x2 WMMA tiles, 4 accumulators):
// 4 fragment loads feed 4 WMMAs per 32-K step (2x operand reuse vs 1x1).
// Main K-loop is branch-free (fast b128 path); guarded tail handles K%32.
// 8 waves / 256-thread block. Batched via gridDim.z: z = b*Hdim + h.
// Requires M%32==0 && N%32==0 (true for every call in this network).
// =====================================================================
__global__ __launch_bounds__(256) void gemm_bf16_nt(
    const float* __restrict__ A, const float* __restrict__ Bw, float* __restrict__ C,
    int M, int N, int K, int lda, int ldb, int ldc,
    int Hdim, long sAb, long sAh, long sBb, long sBh, long sCb, long sCh,
    const float* __restrict__ bias, float alpha, int epi)   // epi: 0 none, 1 bias, 2 bias+relu
{
  const int z  = blockIdx.z;
  const int bI = z / Hdim, hI = z % Hdim;
  A  += (long)bI * sAb + (long)hI * sAh;
  Bw += (long)bI * sBb + (long)hI * sBh;
  C  += (long)bI * sCb + (long)hI * sCh;

  const int wave = threadIdx.x >> 5;
  const int lane = threadIdx.x & 31;
  const int nt   = N >> 5;                          // 32-wide macro-tiles
  const long tile = (long)blockIdx.x * 8 + wave;
  if (tile >= (long)(M >> 5) * nt) return;          // wave-uniform exit (EXEC stays all-1)
  const int tm  = (int)(tile / nt);
  const int tn  = (int)(tile % nt);
  const int l15 = lane & 15;
  const int hi  = lane >> 4;                        // lane half selects K sub-range

  const float* ap0 = A  + (long)(tm * 32 + l15) * lda;   // A rows m, m+16 (K contiguous)
  const float* ap1 = ap0 + (long)16 * lda;
  const float* bp0 = Bw + (long)(tn * 32 + l15) * ldb;   // weight rows n, n+16 (K contiguous)
  const float* bp1 = bp0 + (long)16 * ldb;

  f32x8 acc00 = {}, acc01 = {}, acc10 = {}, acc11 = {};

  const int kMain = K & ~31;                        // unconditionally-fast region
  for (int k0 = 0; k0 < kMain; k0 += 32) {
    const int kA = k0 + hi * 8;
    const int kB = k0 + hi * 8;
    bf16x16 a0 = load_fragA_fast(ap0, kA);
    bf16x16 a1 = load_fragA_fast(ap1, kA);
    bf16x16 b0 = load_fragB_fast(bp0, kB);
    bf16x16 b1 = load_fragB_fast(bp1, kB);
    if (k0 + 64 <= kMain) {                        // software prefetch next K panel
      __builtin_prefetch(ap0 + kA + 32, 0, 1);
      __builtin_prefetch(ap1 + kA + 32, 0, 1);
      __builtin_prefetch(bp0 + kB + 32, 0, 1);
      __builtin_prefetch(bp1 + kB + 32, 0, 1);
    }
    acc00 = __builtin_amdgcn_wmma_f32_16x16x32_bf16(false, a0, false, b0, (short)0, acc00, false, false);
    acc01 = __builtin_amdgcn_wmma_f32_16x16x32_bf16(false, a0, false, b1, (short)0, acc01, false, false);
    acc10 = __builtin_amdgcn_wmma_f32_16x16x32_bf16(false, a1, false, b0, (short)0, acc10, false, false);
    acc11 = __builtin_amdgcn_wmma_f32_16x16x32_bf16(false, a1, false, b1, (short)0, acc11, false, false);
  }
  if (kMain < K) {                                 // single guarded tail step (e.g. conv1, K=6)
    const int kA = kMain + hi * 8;
    const int kB = kMain + hi * 8;
    bf16x16 a0 = load_fragA_guard(ap0, kA, K);
    bf16x16 a1 = load_fragA_guard(ap1, kA, K);
    bf16x16 b0 = load_fragB_guard(bp0, kB, K);
    bf16x16 b1 = load_fragB_guard(bp1, kB, K);
    acc00 = __builtin_amdgcn_wmma_f32_16x16x32_bf16(false, a0, false, b0, (short)0, acc00, false, false);
    acc01 = __builtin_amdgcn_wmma_f32_16x16x32_bf16(false, a0, false, b1, (short)0, acc01, false, false);
    acc10 = __builtin_amdgcn_wmma_f32_16x16x32_bf16(false, a1, false, b0, (short)0, acc10, false, false);
    acc11 = __builtin_amdgcn_wmma_f32_16x16x32_bf16(false, a1, false, b1, (short)0, acc11, false, false);
  }

  // C layout per tile: VGPR r, lane -> row tm*32 + r + 8*hi (+16 for second M tile),
  //                    col tn*32 + l15 (+16 for second N tile)
  const int cn0 = tn * 32 + l15;
  const int cn1 = cn0 + 16;
  const float bv0 = (epi >= 1 && bias) ? bias[cn0] : 0.f;
  const float bv1 = (epi >= 1 && bias) ? bias[cn1] : 0.f;
#pragma unroll
  for (int r = 0; r < 8; ++r) {
    const int cm0 = tm * 32 + r + hi * 8;
    const int cm1 = cm0 + 16;
    float v00 = acc00[r] * alpha + bv0;
    float v01 = acc01[r] * alpha + bv1;
    float v10 = acc10[r] * alpha + bv0;
    float v11 = acc11[r] * alpha + bv1;
    if (epi == 2) {
      v00 = fmaxf(v00, 0.f); v01 = fmaxf(v01, 0.f);
      v10 = fmaxf(v10, 0.f); v11 = fmaxf(v11, 0.f);
    }
    C[(long)cm0 * ldc + cn0] = v00;
    C[(long)cm0 * ldc + cn1] = v01;
    C[(long)cm1 * ldc + cn0] = v10;
    C[(long)cm1 * ldc + cn1] = v11;
  }
}

// ---------------- reductions ----------------
__device__ __forceinline__ float wave_sum(float v) {
#pragma unroll
  for (int o = 16; o > 0; o >>= 1) v += __shfl_down(v, o, 32);
  return v;
}
__device__ __forceinline__ float wave_max(float v) {
#pragma unroll
  for (int o = 16; o > 0; o >>= 1) v = fmaxf(v, __shfl_down(v, o, 32));
  return v;
}
__device__ __forceinline__ float block_sum(float v, float* sh) {
  int lane = threadIdx.x & 31, w = threadIdx.x >> 5;
  v = wave_sum(v);
  if (lane == 0) sh[w] = v;
  __syncthreads();
  float r = (threadIdx.x < 8) ? sh[threadIdx.x] : 0.f;
  if (w == 0) { r = wave_sum(r); if (lane == 0) sh[0] = r; }
  __syncthreads();
  r = sh[0];
  __syncthreads();
  return r;
}
__device__ __forceinline__ float block_max(float v, float* sh) {
  int lane = threadIdx.x & 31, w = threadIdx.x >> 5;
  v = wave_max(v);
  if (lane == 0) sh[w] = v;
  __syncthreads();
  float r = (threadIdx.x < 8) ? sh[threadIdx.x] : -3.4e38f;
  if (w == 0) { r = wave_max(r); if (lane == 0) sh[0] = r; }
  __syncthreads();
  r = sh[0];
  __syncthreads();
  return r;
}

// ---------------- KNN (top-20 by -dist^2, stable ties like lax.top_k) ----------------
__global__ __launch_bounds__(256) void knn_kernel(const float* __restrict__ p, int* __restrict__ idx)
{
  int t = blockIdx.x * blockDim.x + threadIdx.x;
  if (t >= Bn * Nn) return;
  int b = t / Nn, n = t % Nn;
  const float* pb = p + (long)b * 3 * Nn;
  float px = pb[n], py = pb[Nn + n], pz = pb[2 * Nn + n];
  float bv[KNN]; int bi[KNN];
#pragma unroll
  for (int i = 0; i < KNN; ++i) { bv[i] = -3.4e38f; bi[i] = 0; }
  for (int m = 0; m < Nn; ++m) {
    float dx = px - pb[m], dy = py - pb[Nn + m], dz = pz - pb[2 * Nn + m];
    float nd = -(dx * dx + dy * dy + dz * dz);
    if (nd > bv[KNN - 1]) {
      int j = KNN - 1;
      while (j > 0 && nd > bv[j - 1]) { bv[j] = bv[j - 1]; bi[j] = bi[j - 1]; --j; }
      bv[j] = nd; bi[j] = m;
    }
  }
  int* op = idx + (long)t * KNN;
#pragma unroll
  for (int i = 0; i < KNN; ++i) op[i] = bi[i];
}

// ---------------- edge features: [neigh-center, center] (B,N,K,6) ----------------
__global__ void edge_kernel(const float* __restrict__ p, const int* __restrict__ idx, float* __restrict__ e)
{
  long t = (long)blockIdx.x * blockDim.x + threadIdx.x;
  if (t >= MBIG) return;
  long bn = t / KNN;
  int n = (int)(bn % Nn), b = (int)(bn / Nn);
  int j = idx[t];
  const float* pb = p + (long)b * 3 * Nn;
  float cx = pb[n], cy = pb[Nn + n], cz = pb[2 * Nn + n];
  float nx = pb[j], ny = pb[Nn + j], nz = pb[2 * Nn + j];
  float* o = e + t * 6;
  o[0] = nx - cx; o[1] = ny - cy; o[2] = nz - cz;
  o[3] = cx;      o[4] = cy;      o[5] = cz;
}

// ---------------- BatchNorm (batch stats, biased var) ----------------
__global__ __launch_bounds__(256) void bn_stats(const float* __restrict__ y, float* __restrict__ mean,
                                                float* __restrict__ var, long M, int C)
{
  __shared__ float sh[8];
  const int c = blockIdx.x;
  float s = 0.f, q = 0.f;
  for (long i = threadIdx.x; i < M; i += blockDim.x) { float v = y[i * C + c]; s += v; q += v * v; }
  s = block_sum(s, sh);
  q = block_sum(q, sh);
  if (threadIdx.x == 0) { float mu = s / M; mean[c] = mu; var[c] = q / M - mu * mu; }
}
__global__ void bn_apply_relu(const float* __restrict__ y, float* __restrict__ dst,
                              const float* __restrict__ mean, const float* __restrict__ var,
                              const float* __restrict__ g, const float* __restrict__ b,
                              long total, int C)
{
  long t = (long)blockIdx.x * blockDim.x + threadIdx.x;
  if (t >= total) return;
  int c = (int)(t % C);
  float sc = g[c] * rsqrtf(var[c] + 1e-5f);
  float v = (y[t] - mean[c]) * sc + b[c];
  dst[t] = fmaxf(v, 0.f);
}

// ---------------- max-pool over K neighbors ----------------
__global__ void pool_max(const float* __restrict__ x, float* __restrict__ dst, int C, int ldout)
{
  long t = (long)blockIdx.x * blockDim.x + threadIdx.x;
  long total = (long)BNr * C;
  if (t >= total) return;
  int c = (int)(t % C); long bn = t / C;
  const float* xp = x + bn * KNN * (long)C + c;
  float m = xp[0];
#pragma unroll
  for (int k = 1; k < KNN; ++k) m = fmaxf(m, xp[(long)k * C]);
  dst[bn * ldout + c] = m;
}

// ---------------- LayerNorm (ddof=1, eps added to std) on rows of 512 ----------------
__global__ __launch_bounds__(256) void layernorm_rows(const float* __restrict__ x, float* __restrict__ y,
                                                      const float* __restrict__ g, const float* __restrict__ b)
{
  __shared__ float sh[8];
  const long row = blockIdx.x;
  const float* xr = x + row * Dn;
  float s = 0.f, q = 0.f;
  for (int i = threadIdx.x; i < Dn; i += blockDim.x) { float v = xr[i]; s += v; q += v * v; }
  s = block_sum(s, sh);
  q = block_sum(q, sh);
  const float mean = s / Dn;
  const float var  = (q - Dn * mean * mean) / (Dn - 1);
  const float inv  = 1.f / (sqrtf(fmaxf(var, 0.f)) + 1e-6f);
  float* yr = y + row * Dn;
  for (int i = threadIdx.x; i < Dn; i += blockDim.x)
    yr[i] = g[i] * (xr[i] - mean) * inv + b[i];
}

// ---------------- stride-generalized softmax (rows / columns) ----------------
__global__ __launch_bounds__(256) void softmax_rows(const float* __restrict__ in, float* __restrict__ out,
    int L, int rowsPerBatch, long bsi, long rsi, long esi, long bso, long rso, long eso)
{
  __shared__ float sh[8];
  const long row = blockIdx.x;
  const long bI = row / rowsPerBatch, r = row % rowsPerBatch;
  const float* ip = in + bI * bsi + r * rsi;
  float* op = out + bI * bso + r * rso;
  float m = -3.4e38f;
  for (int i = threadIdx.x; i < L; i += blockDim.x) m = fmaxf(m, ip[(long)i * esi]);
  m = block_max(m, sh);
  float s = 0.f;
  for (int i = threadIdx.x; i < L; i += blockDim.x) s += expf(ip[(long)i * esi] - m);
  s = block_sum(s, sh);
  const float invs = 1.f / s;
  for (int i = threadIdx.x; i < L; i += blockDim.x) op[(long)i * eso] = expf(ip[(long)i * esi] - m) * invs;
}

// ---------------- misc elementwise ----------------
__global__ void add_kernel(const float* __restrict__ a, const float* __restrict__ b,
                           float* __restrict__ c, long n)
{
  long t = (long)blockIdx.x * blockDim.x + threadIdx.x;
  if (t < n) c[t] = a[t] + b[t];
}
// v:(B,N,H*dk) -> vT:(B,H,dk,N)  so A@vT^T via the NT GEMM performs attn@v
__global__ void transpose_heads(const float* __restrict__ v, float* __restrict__ vT)
{
  long t = (long)blockIdx.x * blockDim.x + threadIdx.x;
  const long total = (long)Bn * Hn * DKn * Nn;
  if (t >= total) return;
  int n = (int)(t % Nn); long r = t / Nn;
  int d = (int)(r % DKn); r /= DKn;
  int h = (int)(r % Hn);  int b = (int)(r / Hn);
  vT[t] = v[((long)(b * Nn + n)) * Dn + h * DKn + d];
}

// =====================================================================
// Host-side orchestration
// =====================================================================
static void gemmH(hipStream_t st, const float* A, const float* Bw, float* C,
                  int M, int N, int K, int lda, int ldb, int ldc,
                  int batches, int Hdim, long sAb, long sAh, long sBb, long sBh, long sCb, long sCh,
                  const float* bias, float alpha, int epi)
{
  long tiles = (long)(M / 32) * (N / 32);
  dim3 grid((unsigned)((tiles + 7) / 8), 1, (unsigned)batches);
  gemm_bf16_nt<<<grid, 256, 0, st>>>(A, Bw, C, M, N, K, lda, ldb, ldc,
                                     Hdim, sAb, sAh, sBb, sBh, sCb, sCh, bias, alpha, epi);
}

struct TS {  // transformer scratch (carved from arena2)
  float *xn, *q, *k, *v, *o, *vT, *mo, *x1, *ffh, *attn;
};

static void mhaH(hipStream_t st, const float* qx, const float* kx, const float* vx,
                 const float* W, const float* Bias, float* out, const TS& S)
{
  gemmH(st, qx, W + 0 * Dn * Dn, S.q, BNr, Dn, Dn, Dn, Dn, Dn, 1, 1, 0,0,0,0,0,0, Bias + 0 * Dn, 1.f, 1);
  gemmH(st, kx, W + 1 * Dn * Dn, S.k, BNr, Dn, Dn, Dn, Dn, Dn, 1, 1, 0,0,0,0,0,0, Bias + 1 * Dn, 1.f, 1);
  gemmH(st, vx, W + 2 * Dn * Dn, S.v, BNr, Dn, Dn, Dn, Dn, Dn, 1, 1, 0,0,0,0,0,0, Bias + 2 * Dn, 1.f, 1);
  long ttot = (long)Bn * Hn * DKn * Nn;
  transpose_heads<<<(unsigned)((ttot + 255) / 256), 256, 0, st>>>(S.v, S.vT);
  // attn = (q @ k^T) / sqrt(dk), per (b,h)
  gemmH(st, S.q, S.k, S.attn, Nn, Nn, DKn, Dn, Dn, Nn,
        Bn * Hn, Hn, (long)Nn * Dn, DKn, (long)Nn * Dn, DKn, (long)Hn * Nn * Nn, (long)Nn * Nn,
        nullptr, 0.08838834764831845f, 0);
  softmax_rows<<<Bn * Hn * Nn, 256, 0, st>>>(S.attn, S.attn, Nn, Bn * Hn * Nn,
                                             0L, (long)Nn, 1L, 0L, (long)Nn, 1L);
  // o = attn @ v   (B-operand = vT, K-contiguous)
  gemmH(st, S.attn, S.vT, S.o, Nn, DKn, Nn, Nn, Nn, Dn,
        Bn * Hn, Hn, (long)Hn * Nn * Nn, (long)Nn * Nn, (long)Hn * DKn * Nn, (long)DKn * Nn,
        (long)Nn * Dn, DKn, nullptr, 1.f, 0);
  gemmH(st, S.o, W + 3 * Dn * Dn, out, BNr, Dn, Dn, Dn, Dn, Dn, 1, 1, 0,0,0,0,0,0, Bias + 3 * Dn, 1.f, 1);
}

static void ffnH(hipStream_t st, const float* xn, const float* w1, const float* b1,
                 const float* w2, const float* b2, float* ffh, float* out)
{
  gemmH(st, xn,  w1, ffh, BNr, FFn, Dn,  Dn,  Dn,  FFn, 1, 1, 0,0,0,0,0,0, b1, 1.f, 2); // bias+relu
  gemmH(st, ffh, w2, out, BNr, Dn,  FFn, FFn, FFn, Dn,  1, 1, 0,0,0,0,0,0, b2, 1.f, 1);
}

static void encodeH(hipStream_t st, const float* src, float* mem,
                    const float* aw, const float* ab,
                    const float* fw1, const float* fb1, const float* fw2, const float* fb2,
                    const float* ng, const float* nb, const TS& S)
{
  const long NE = (long)BNr * Dn;
  layernorm_rows<<<BNr, 256, 0, st>>>(src, S.xn, ng + 0, nb + 0);
  mhaH(st, S.xn, S.xn, S.xn, aw, ab, S.mo, S);
  add_kernel<<<(unsigned)((NE + 255) / 256), 256, 0, st>>>(src, S.mo, S.x1, NE);
  layernorm_rows<<<BNr, 256, 0, st>>>(S.x1, S.xn, ng + Dn, nb + Dn);
  ffnH(st, S.xn, fw1, fb1, fw2, fb2, S.ffh, S.mo);
  add_kernel<<<(unsigned)((NE + 255) / 256), 256, 0, st>>>(S.x1, S.mo, S.x1, NE);
  layernorm_rows<<<BNr, 256, 0, st>>>(S.x1, mem, ng + 2 * Dn, nb + 2 * Dn);
}

static void decodeH(hipStream_t st, const float* tgt, const float* mem, float* out,
                    const float* sw, const float* sb, const float* xw, const float* xb,
                    const float* fw1, const float* fb1, const float* fw2, const float* fb2,
                    const float* ng, const float* nb, const TS& S)
{
  const long NE = (long)BNr * Dn;
  layernorm_rows<<<BNr, 256, 0, st>>>(tgt, S.xn, ng + 0, nb + 0);
  mhaH(st, S.xn, S.xn, S.xn, sw, sb, S.mo, S);
  add_kernel<<<(unsigned)((NE + 255) / 256), 256, 0, st>>>(tgt, S.mo, S.x1, NE);
  layernorm_rows<<<BNr, 256, 0, st>>>(S.x1, S.xn, ng + Dn, nb + Dn);
  mhaH(st, S.xn, mem, mem, xw, xb, S.mo, S);
  add_kernel<<<(unsigned)((NE + 255) / 256), 256, 0, st>>>(S.x1, S.mo, S.x1, NE);
  layernorm_rows<<<BNr, 256, 0, st>>>(S.x1, S.xn, ng + 2 * Dn, nb + 2 * Dn);
  ffnH(st, S.xn, fw1, fb1, fw2, fb2, S.ffh, S.mo);
  add_kernel<<<(unsigned)((NE + 255) / 256), 256, 0, st>>>(S.x1, S.mo, S.x1, NE);
  layernorm_rows<<<BNr, 256, 0, st>>>(S.x1, out, ng + 3 * Dn, nb + 3 * Dn);
}

static void dgcnnH(hipStream_t st, const float* p, int* idx, float* edge,
                   float* A1, float* A2, float* mean, float* var,
                   const float* const* cw, const float* const* bng, const float* const* bnb,
                   float* cat, float* feat)
{
  knn_kernel<<<(Bn * Nn + 255) / 256, 256, 0, st>>>(p, idx);
  edge_kernel<<<(unsigned)((MBIG + 255) / 256), 256, 0, st>>>(p, idx, edge);
  const int cin[4]  = {6, 64, 64, 128};
  const int cout[4] = {64, 64, 128, 256};
  const int coff[4] = {0, 64, 128, 256};
  const float* x = edge; int lda = 6;
  float* bufs[2] = {A1, A2};
  for (int L = 0; L < 4; ++L) {
    float* y = bufs[L & 1];
    gemmH(st, x, cw[L], y, (int)MBIG, cout[L], cin[L], lda, cin[L], cout[L],
          1, 1, 0,0,0,0,0,0, nullptr, 1.f, 0);
    bn_stats<<<cout[L], 256, 0, st>>>(y, mean, var, MBIG, cout[L]);
    long tot = MBIG * cout[L];
    bn_apply_relu<<<(unsigned)((tot + 255) / 256), 256, 0, st>>>(y, y, mean, var, bng[L], bnb[L], tot, cout[L]);
    long ptot = (long)BNr * cout[L];
    pool_max<<<(unsigned)((ptot + 255) / 256), 256, 0, st>>>(y, cat + coff[L], cout[L], 512);
    x = y; lda = cout[L];
  }
  // conv5: 512 -> 512 on pooled concat, BN+ReLU -> feat
  gemmH(st, cat, cw[4], A1, BNr, 512, 512, 512, 512, 512, 1, 1, 0,0,0,0,0,0, nullptr, 1.f, 0);
  bn_stats<<<512, 256, 0, st>>>(A1, mean, var, (long)BNr, 512);
  long tot = (long)BNr * 512;
  bn_apply_relu<<<(unsigned)((tot + 255) / 256), 256, 0, st>>>(A1, feat, mean, var, bng[4], bnb[4], tot, 512);
}

extern "C" void kernel_launch(void* const* d_in, const int* in_sizes, int n_in,
                              void* d_out, int out_size, void* d_ws, size_t ws_size,
                              hipStream_t stream)
{
  (void)in_sizes; (void)n_in; (void)out_size; (void)ws_size;
  const float* ps = (const float*)d_in[0];
  const float* pt = (const float*)d_in[1];
  const float* cw[5]  = {(const float*)d_in[2], (const float*)d_in[3], (const float*)d_in[4],
                         (const float*)d_in[5], (const float*)d_in[6]};
  const float* bng[5] = {(const float*)d_in[7],  (const float*)d_in[9],  (const float*)d_in[11],
                         (const float*)d_in[13], (const float*)d_in[15]};
  const float* bnb[5] = {(const float*)d_in[8],  (const float*)d_in[10], (const float*)d_in[12],
                         (const float*)d_in[14], (const float*)d_in[16]};
  const float* enc_attn_w = (const float*)d_in[17];
  const float* enc_attn_b = (const float*)d_in[18];
  const float* enc_ff_w1  = (const float*)d_in[19];
  const float* enc_ff_b1  = (const float*)d_in[20];
  const float* enc_ff_w2  = (const float*)d_in[21];
  const float* enc_ff_b2  = (const float*)d_in[22];
  const float* enc_ng     = (const float*)d_in[23];
  const float* enc_nb     = (const float*)d_in[24];
  const float* dec_self_w = (const float*)d_in[25];
  const float* dec_self_b = (const float*)d_in[26];
  const float* dec_src_w  = (const float*)d_in[27];
  const float* dec_src_b  = (const float*)d_in[28];
  const float* dec_ff_w1  = (const float*)d_in[29];
  const float* dec_ff_b1  = (const float*)d_in[30];
  const float* dec_ff_w2  = (const float*)d_in[31];
  const float* dec_ff_b2  = (const float*)d_in[32];
  const float* dec_ng     = (const float*)d_in[33];
  const float* dec_nb     = (const float*)d_in[34];
  float* out = (float*)d_out;

  // ---- workspace bump allocator ----
  char* base = (char*)d_ws;
  size_t off = 0;
  auto alloc = [&](size_t nbytes) -> void* {
    void* r = base + off;
    off += (nbytes + 255) & ~(size_t)255;
    return r;
  };
  const long SL = (long)BNr * Dn;                         // 2,097,152 floats
  int*   idx  = (int*)  alloc(MBIG * sizeof(int));
  float* edge = (float*)alloc(MBIG * 6 * sizeof(float));
  float* A1   = (float*)alloc((size_t)Bn * Hn * Nn * Nn * sizeof(float));  // 16.78M f: y1/y3, attn, m
  float* A2   = (float*)alloc((size_t)10 * SL * sizeof(float));            // 20.97M f: y2/y4, TS carve
  float* mean = (float*)alloc(512 * sizeof(float));
  float* var  = (float*)alloc(512 * sizeof(float));
  float* cat_s = (float*)alloc(SL * sizeof(float));
  float* cat_t = (float*)alloc(SL * sizeof(float));
  float* fs    = (float*)alloc(SL * sizeof(float));
  float* ft    = (float*)alloc(SL * sizeof(float));
  float* mem_s = (float*)alloc(SL * sizeof(float));
  float* mem_t = (float*)alloc(SL * sizeof(float));
  float* tgt_e = (float*)alloc(SL * sizeof(float));
  float* src_e = (float*)alloc(SL * sizeof(float));
  float* ps_e  = (float*)alloc(SL * sizeof(float));
  float* pt_e  = (float*)alloc(SL * sizeof(float));

  // ---- DGCNN features for both clouds ----
  dgcnnH(stream, ps, idx, edge, A1, A2, mean, var, cw, bng, bnb, cat_s, fs);
  dgcnnH(stream, pt, idx, edge, A1, A2, mean, var, cw, bng, bnb, cat_t, ft);

  // ---- transformer scratch carved from arena2 (exactly 10*SL), attn in arena1 ----
  TS S;
  S.xn  = A2 + 0 * SL;  S.q  = A2 + 1 * SL;  S.k   = A2 + 2 * SL;  S.v  = A2 + 3 * SL;
  S.o   = A2 + 4 * SL;  S.vT = A2 + 5 * SL;  S.mo  = A2 + 6 * SL;  S.x1 = A2 + 7 * SL;
  S.ffh = A2 + 8 * SL;  S.attn = A1;

  // tgt_emb = decode(ft, encode(fs)); src_emb = decode(fs, encode(ft))
  encodeH(stream, fs, mem_s, enc_attn_w, enc_attn_b, enc_ff_w1, enc_ff_b1, enc_ff_w2, enc_ff_b2,
          enc_ng, enc_nb, S);
  decodeH(stream, ft, mem_s, tgt_e, dec_self_w, dec_self_b, dec_src_w, dec_src_b,
          dec_ff_w1, dec_ff_b1, dec_ff_w2, dec_ff_b2, dec_ng, dec_nb, S);
  encodeH(stream, ft, mem_t, enc_attn_w, enc_attn_b, enc_ff_w1, enc_ff_b1, enc_ff_w2, enc_ff_b2,
          enc_ng, enc_nb, S);
  decodeH(stream, fs, mem_t, src_e, dec_self_w, dec_self_b, dec_src_w, dec_src_b,
          dec_ff_w1, dec_ff_b1, dec_ff_w2, dec_ff_b2, dec_ng, dec_nb, S);

  const long NE = (long)BNr * Dn;
  add_kernel<<<(unsigned)((NE + 255) / 256), 256, 0, stream>>>(fs, src_e, ps_e, NE);
  add_kernel<<<(unsigned)((NE + 255) / 256), 256, 0, stream>>>(ft, tgt_e, pt_e, NE);

  // m = ps_emb @ pt_emb^T / sqrt(D), per batch, into arena1
  float* m = A1;
  gemmH(stream, ps_e, pt_e, m, Nn, Nn, Dn, Dn, Dn, Nn,
        Bn, 1, (long)Nn * Dn, 0, (long)Nn * Dn, 0, (long)Nn * Nn, 0,
        nullptr, 0.044194173824159216f, 0);

  // out[0] = softmax over axis 2 (rows); out[1] = softmax over axis 1 (columns)
  const long NNl = (long)Nn * Nn;
  softmax_rows<<<BNr, 256, 0, stream>>>(m, out,             Nn, Nn, NNl, (long)Nn, 1L, NNl, (long)Nn, 1L);
  softmax_rows<<<BNr, 256, 0, stream>>>(m, out + Bn * NNl,  Nn, Nn, NNl, 1L, (long)Nn, NNl, 1L, (long)Nn);
}
